// MambaGuidedAttentionWrapper_57724360458715
// MI455X (gfx1250) — matrix operand order repaired and checked by reading, not compiled, not run on hardware
//
#include <hip/hip_runtime.h>
#include <math.h>

// ---------------- constants (match reference) ----------------
constexpr int Bc = 2, Lc = 2048, Dc = 2048, Hc = 16, DHc = 128, DRELc = 64;
constexpr int KTOP = 512;                          // 0.25 * L
constexpr float SCALE_ATTN = 0.08838834764831845f; // DH^-0.5 = 128^-0.5
constexpr float SCALE_REL  = 0.125f;               // D_REL^-0.5 = 64^-0.5

typedef __attribute__((ext_vector_type(16))) __bf16 v16bf;
typedef __attribute__((ext_vector_type(8)))  float  v8f;

union Frag { v16bf v; uint4 u[2]; };

static __device__ __forceinline__ unsigned short f2bf(float f) {
  unsigned int x = __float_as_uint(f);
  unsigned int r = x + 0x7fffu + ((x >> 16) & 1u);   // round-to-nearest-even
  return (unsigned short)(r >> 16);
}
static __device__ __forceinline__ unsigned int sortkey(float f) {
  unsigned int u = __float_as_uint(f);
  return (u & 0x80000000u) ? ~u : (u | 0x80000000u); // monotone float->uint
}

// ---------------- f32 -> bf16 conversion ----------------
__global__ __launch_bounds__(256) void cvt_f32_bf16(const float* __restrict__ in,
                                                    unsigned short* __restrict__ out, int n) {
  int i = blockIdx.x * 256 + threadIdx.x;
  if (i < n) out[i] = f2bf(in[i]);
}

// ---------------- generic WMMA GEMM: C[M,N] = scale * A[M,K] * W[N,K]^T ----------------
// A, W bf16 row-major. 4 waves/block; each wave does a (MT*16) x (NT*16) tile.
// Per k-step: 2*MT A-loads + 2*NT B-loads feed MT*NT WMMAs.
enum { OUT_F32 = 0, OUT_BF16 = 1, OUT_BF16_BT = 2 }; // BT: per-batch transposed bf16 (for V)

template<int M, int N, int KD, int MT, int NT, int OM>
__global__ __launch_bounds__(128) void gemm_wmma(const unsigned short* __restrict__ A,
                                                 const unsigned short* __restrict__ W,
                                                 void* __restrict__ Cptr, float scale) {
  const int lane = threadIdx.x & 31, wave = threadIdx.x >> 5;
  const int hh = lane >> 4, nn = lane & 15;
  const int row0 = blockIdx.y * (MT * 16);
  const int col0 = blockIdx.x * (NT * 64) + wave * (NT * 16);
  v8f acc[MT * NT] = {};
  const unsigned short* arow[MT];
#pragma unroll
  for (int mi = 0; mi < MT; ++mi)
    arow[mi] = A + (size_t)(row0 + mi * 16 + nn) * KD;
  const unsigned short* wrow[NT];
#pragma unroll
  for (int t = 0; t < NT; ++t)
    wrow[t] = W + (size_t)(col0 + t * 16 + nn) * KD;

  for (int k0 = 0; k0 < KD; k0 += 32) {
    Frag a[MT];  // A 16x32 layout: lane(h,m): K = h*16+{0..15}
#pragma unroll
    for (int mi = 0; mi < MT; ++mi) {
      a[mi].u[0] = *(const uint4*)(arow[mi] + k0 + hh * 16);
      a[mi].u[1] = *(const uint4*)(arow[mi] + k0 + hh * 16 + 8);
      if (k0 + 256 < KD) __builtin_prefetch(arow[mi] + k0 + 256, 0, 3);
    }
#pragma unroll
    for (int t = 0; t < NT; ++t) {
      Frag b;  // B 32x16 layout: lane(h,n): K = h*16+{0..15} of column n == W row n
      b.u[0] = *(const uint4*)(wrow[t] + k0 + hh * 16);
      b.u[1] = *(const uint4*)(wrow[t] + k0 + hh * 16 + 8);
      if (k0 + 256 < KD) __builtin_prefetch(wrow[t] + k0 + 256, 0, 3);
#pragma unroll
      for (int mi = 0; mi < MT; ++mi)
        acc[mi * NT + t] = __builtin_amdgcn_wmma_f32_16x16x32_bf16(
            false, a[mi].v, false, b.v, (short)0, acc[mi * NT + t], false, false);
    }
  }
#pragma unroll
  for (int mi = 0; mi < MT; ++mi) {
#pragma unroll
    for (int t = 0; t < NT; ++t) {
#pragma unroll
      for (int r = 0; r < 8; ++r) {  // C layout: VGPR r -> row r+8h, col = lane%16
        const int row = row0 + mi * 16 + r + 8 * hh;
        const int col = col0 + t * 16 + nn;
        const float val = acc[mi * NT + t][r] * scale;
        if (OM == OUT_F32) {
          ((float*)Cptr)[(size_t)row * N + col] = val;
        } else if (OM == OUT_BF16) {
          ((unsigned short*)Cptr)[(size_t)row * N + col] = f2bf(val);
        } else { // row = b*Lc + l  ->  Vt[(b*Dc + col)*Lc + l]
          const int bb = row / Lc, l = row % Lc;
          ((unsigned short*)Cptr)[((size_t)bb * Dc + col) * Lc + l] = f2bf(val);
        }
      }
    }
  }
}

// ---------------- top-K threshold + bitmask (one block per (b,q) row) ----------------
__global__ __launch_bounds__(256) void topk_mask_kernel(const float* __restrict__ Srel,
                                                        unsigned int* __restrict__ maskbits) {
  __shared__ unsigned int skey[Lc];
  __shared__ unsigned int cnt;
  const int bq = blockIdx.x;
  const int q = bq & (Lc - 1);
  const int tid = threadIdx.x;
  const float* row = Srel + (size_t)bq * Lc;
  const int valid = q + 1;                    // causal: keys 0..q
  for (int i = tid; i < Lc; i += 256)
    skey[i] = (i < valid) ? sortkey(row[i]) : 0u;   // key 0 < any finite score key
  __syncthreads();
  unsigned int thr = 0u;
  if (valid > KTOP) {
    unsigned int lo = 0u, hi = 0xFFFFFFFFu;   // find largest t with count(key>=t) >= K
    while (lo < hi) {
      const unsigned int mid = lo + ((hi - lo) >> 1) + 1u;
      if (tid == 0) cnt = 0u;
      __syncthreads();
      unsigned int c = 0;
      for (int i = tid; i < Lc; i += 256) c += (skey[i] >= mid) ? 1u : 0u;
      atomicAdd(&cnt, c);
      __syncthreads();
      const unsigned int total = cnt;
      __syncthreads();
      if (total >= (unsigned)KTOP) lo = mid; else hi = mid - 1u;
    }
    thr = lo;
  }
  for (int w = tid; w < Lc / 32; w += 256) {
    unsigned int word = 0u;
    for (int j = 0; j < 32; ++j) {
      const int k = w * 32 + j;
      const bool ok = (k < q) ? ((valid <= KTOP) || (skey[k] >= thr)) : (k == q);
      word |= (ok ? 1u : 0u) << j;
    }
    maskbits[(size_t)bq * (Lc / 32) + w] = word;
  }
}

// ---------------- fused flash attention with precomputed bitmask ----------------
// grid: (L/64, H, B); 4 waves/block, one 16-query tile per wave.
__global__ __launch_bounds__(128) void attn_kernel(const unsigned short* __restrict__ Qb,
                                                   const unsigned short* __restrict__ Kb,
                                                   const unsigned short* __restrict__ Vt,
                                                   const unsigned int* __restrict__ maskbits,
                                                   unsigned short* __restrict__ ctx) {
  __shared__ __align__(16) unsigned short plds[4][16 * 32];  // per-wave P staging tile
  const int lane = threadIdx.x & 31, wave = threadIdx.x >> 5;
  const int hh = lane >> 4, nn = lane & 15;
  const int b = blockIdx.z, hd = blockIdx.y;
  const int q0 = (blockIdx.x * 4 + wave) * 16;
  const size_t headBase = (size_t)b * Lc * Dc + (size_t)hd * DHc;

  Frag qf[4];  // Q fragments for the 4 feature steps of 32 (DH=128)
  {
    const unsigned short* qrow = Qb + headBase + (size_t)(q0 + nn) * Dc;
#pragma unroll
    for (int fs = 0; fs < 4; ++fs) {
      qf[fs].u[0] = *(const uint4*)(qrow + fs * 32 + hh * 16);
      qf[fs].u[1] = *(const uint4*)(qrow + fs * 32 + hh * 16 + 8);
    }
  }
  v8f oacc[8] = {};                 // 16 q x 128 feat f32 accumulator
  float rowmax[8], rowsum[8];
#pragma unroll
  for (int r = 0; r < 8; ++r) { rowmax[r] = -1e30f; rowsum[r] = 0.0f; }

  const int ntiles = (q0 + 16 + 31) / 32;   // key tiles of 32 covering [0, q0+15]
  for (int kt = 0; kt < ntiles; ++kt) {
    const int k0 = kt * 32;
    // prefetch next key tile rows
    if (kt + 1 < ntiles) {
      __builtin_prefetch(Kb + headBase + (size_t)(k0 + 32 + nn) * Dc, 0, 3);
      __builtin_prefetch(Kb + headBase + (size_t)(k0 + 48 + nn) * Dc, 0, 3);
    }
    v8f s0 = {}, s1 = {};
#pragma unroll
    for (int fs = 0; fs < 4; ++fs) {
      Frag bk;  // B = K^T chunk; column n = key token -> contiguous K row read
      const unsigned short* kr0 = Kb + headBase + (size_t)(k0 + nn) * Dc + fs * 32 + hh * 16;
      bk.u[0] = *(const uint4*)(kr0);
      bk.u[1] = *(const uint4*)(kr0 + 8);
      s0 = __builtin_amdgcn_wmma_f32_16x16x32_bf16(false, qf[fs].v, false, bk.v,
                                                   (short)0, s0, false, false);
      const unsigned short* kr1 = kr0 + (size_t)16 * Dc;
      bk.u[0] = *(const uint4*)(kr1);
      bk.u[1] = *(const uint4*)(kr1 + 8);
      s1 = __builtin_amdgcn_wmma_f32_16x16x32_bf16(false, qf[fs].v, false, bk.v,
                                                   (short)0, s1, false, false);
    }
    // mask + online softmax (row m = r + 8*hh lives across lanes nn=0..15 of half hh)
    float p0[8], p1[8], corr[8];
#pragma unroll
    for (int r = 0; r < 8; ++r) {
      const int qrow = q0 + r + 8 * hh;
      const unsigned int word =
          maskbits[((size_t)b * Lc + qrow) * (Lc / 32) + (k0 >> 5)];
      const bool ok0 = (word >> nn) & 1u;
      const bool ok1 = (word >> (16 + nn)) & 1u;
      const float v0 = ok0 ? s0[r] * SCALE_ATTN : -1e30f;
      const float v1 = ok1 ? s1[r] * SCALE_ATTN : -1e30f;
      float m = fmaxf(v0, v1);
#pragma unroll
      for (int x = 1; x < 16; x <<= 1) m = fmaxf(m, __shfl_xor(m, x, 32));
      const float mnew = fmaxf(rowmax[r], m);
      const float c = __expf(rowmax[r] - mnew);
      rowmax[r] = mnew;
      corr[r] = c;
      const float e0 = ok0 ? __expf(v0 - mnew) : 0.0f;
      const float e1 = ok1 ? __expf(v1 - mnew) : 0.0f;
      p0[r] = e0; p1[r] = e1;
      rowsum[r] = rowsum[r] * c + e0 + e1;   // per-lane partial, reduced at end
    }
#pragma unroll
    for (int t = 0; t < 8; ++t)
#pragma unroll
      for (int r = 0; r < 8; ++r) oacc[t][r] = oacc[t][r] * corr[r];
    // C-layout P  -> LDS -> A-layout fragment (same-wave DS ops are in-order)
#pragma unroll
    for (int r = 0; r < 8; ++r) {
      const int rr = r + 8 * hh;
      plds[wave][rr * 32 + nn]      = f2bf(p0[r]);
      plds[wave][rr * 32 + 16 + nn] = f2bf(p1[r]);
    }
    asm volatile("s_wait_dscnt 0" ::: "memory");
    Frag pa;
    pa.u[0] = *(const uint4*)&plds[wave][nn * 32 + hh * 16];
    pa.u[1] = *(const uint4*)&plds[wave][nn * 32 + hh * 16 + 8];
    // P (16x32) @ V (32x128): B frags contiguous thanks to transposed V
#pragma unroll
    for (int t = 0; t < 8; ++t) {
      Frag bv;
      const unsigned short* vrow =
          Vt + ((size_t)b * Dc + hd * DHc + t * 16 + nn) * Lc + k0 + hh * 16;
      bv.u[0] = *(const uint4*)(vrow);
      bv.u[1] = *(const uint4*)(vrow + 8);
      oacc[t] = __builtin_amdgcn_wmma_f32_16x16x32_bf16(false, pa.v, false, bv.v,
                                                        (short)0, oacc[t], false, false);
    }
  }
  // normalize and emit bf16 context
  float inv[8];
#pragma unroll
  for (int r = 0; r < 8; ++r) {
    float t = rowsum[r];
#pragma unroll
    for (int x = 1; x < 16; x <<= 1) t += __shfl_xor(t, x, 32);
    inv[r] = 1.0f / t;   // diagonal always unmasked -> t > 0
  }
#pragma unroll
  for (int t = 0; t < 8; ++t)
#pragma unroll
    for (int r = 0; r < 8; ++r)
      ctx[headBase + (size_t)(q0 + r + 8 * hh) * Dc + t * 16 + nn] =
          f2bf(oacc[t][r] * inv[r]);
}

// ---------------- host orchestration ----------------
extern "C" void kernel_launch(void* const* d_in, const int* in_sizes, int n_in,
                              void* d_out, int out_size, void* d_ws, size_t ws_size,
                              hipStream_t stream) {
  (void)in_sizes; (void)n_in; (void)out_size; (void)ws_size;
  const float* hs  = (const float*)d_in[0];
  const float* rel = (const float*)d_in[1];
  const float* wqr = (const float*)d_in[2];
  const float* wkr = (const float*)d_in[3];
  const float* wq  = (const float*)d_in[4];
  const float* wk  = (const float*)d_in[5];
  const float* wv  = (const float*)d_in[6];
  const float* wo  = (const float*)d_in[7];

  char* ws = (char*)d_ws;
  size_t off = 0;
  auto take = [&](size_t bytes) -> char* {
    char* p = ws + off;
    off += (bytes + 255) & ~(size_t)255;
    return p;
  };
  unsigned short* hsb   = (unsigned short*)take((size_t)Bc * Lc * Dc * 2);
  unsigned short* relb  = (unsigned short*)take((size_t)Bc * Lc * Dc * 2);
  unsigned short* wqb   = (unsigned short*)take((size_t)Dc * Dc * 2);
  unsigned short* wkb   = (unsigned short*)take((size_t)Dc * Dc * 2);
  unsigned short* wvb   = (unsigned short*)take((size_t)Dc * Dc * 2);
  unsigned short* wob   = (unsigned short*)take((size_t)Dc * Dc * 2);
  unsigned short* wqrb  = (unsigned short*)take((size_t)DRELc * Dc * 2);
  unsigned short* wkrb  = (unsigned short*)take((size_t)DRELc * Dc * 2);
  unsigned short* Qbuf  = (unsigned short*)take((size_t)Bc * Lc * Dc * 2);
  unsigned short* Kbuf  = (unsigned short*)take((size_t)Bc * Lc * Dc * 2);
  unsigned short* Vt    = (unsigned short*)take((size_t)Bc * Dc * Lc * 2);  // transposed
  unsigned short* qrelb = (unsigned short*)take((size_t)Bc * Lc * DRELc * 2);
  unsigned short* krelb = (unsigned short*)take((size_t)Bc * Lc * DRELc * 2);
  float*          Srel  = (float*)take((size_t)Bc * Lc * Lc * 4);
  unsigned int*   maskb = (unsigned int*)take((size_t)Bc * Lc * (Lc / 32) * 4);
  unsigned short* ctxb  = (unsigned short*)take((size_t)Bc * Lc * Dc * 2);

  auto cvt = [&](const float* in, unsigned short* out, size_t n) {
    cvt_f32_bf16<<<dim3((unsigned)((n + 255) / 256)), dim3(256), 0, stream>>>(in, out, (int)n);
  };
  cvt(hs,  hsb,  (size_t)Bc * Lc * Dc);
  cvt(rel, relb, (size_t)Bc * Lc * Dc);
  cvt(wq,  wqb,  (size_t)Dc * Dc);
  cvt(wk,  wkb,  (size_t)Dc * Dc);
  cvt(wv,  wvb,  (size_t)Dc * Dc);
  cvt(wo,  wob,  (size_t)Dc * Dc);
  cvt(wqr, wqrb, (size_t)DRELc * Dc);
  cvt(wkr, wkrb, (size_t)DRELc * Dc);

  // Q/K/V projections: 32x64 wave tile (MT=2, NT=4) -> 8 WMMAs / 12 loads per k-step
  gemm_wmma<Bc * Lc, Dc, Dc, 2, 4, OUT_BF16>
      <<<dim3(Dc / 256, (Bc * Lc) / 32), 128, 0, stream>>>(hsb, wqb, Qbuf, 1.0f);
  gemm_wmma<Bc * Lc, Dc, Dc, 2, 4, OUT_BF16>
      <<<dim3(Dc / 256, (Bc * Lc) / 32), 128, 0, stream>>>(hsb, wkb, Kbuf, 1.0f);
  gemm_wmma<Bc * Lc, Dc, Dc, 2, 4, OUT_BF16_BT>
      <<<dim3(Dc / 256, (Bc * Lc) / 32), 128, 0, stream>>>(hsb, wvb, Vt, 1.0f);

  // relevance projections (N=64 -> NT=1)
  gemm_wmma<Bc * Lc, DRELc, Dc, 2, 1, OUT_BF16>
      <<<dim3(1, (Bc * Lc) / 32), 128, 0, stream>>>(relb, wqrb, qrelb, 1.0f);
  gemm_wmma<Bc * Lc, DRELc, Dc, 2, 1, OUT_BF16>
      <<<dim3(1, (Bc * Lc) / 32), 128, 0, stream>>>(relb, wkrb, krelb, 1.0f);

  // batched relevance score matrices (scaled by D_REL^-0.5)
  for (int b = 0; b < Bc; ++b)
    gemm_wmma<Lc, Lc, DRELc, 2, 4, OUT_F32>
        <<<dim3(Lc / 256, Lc / 32), 128, 0, stream>>>(
            qrelb + (size_t)b * Lc * DRELc, krelb + (size_t)b * Lc * DRELc,
            Srel + (size_t)b * Lc * Lc, SCALE_REL);

  // top-K threshold + causal/diag bitmask
  topk_mask_kernel<<<dim3(Bc * Lc), 256, 0, stream>>>(Srel, maskb);

  // fused masked flash attention
  attn_kernel<<<dim3(Lc / 64, Hc, Bc), 128, 0, stream>>>(Qbuf, Kbuf, Vt, maskb, ctxb);

  // output projection -> f32 result
  gemm_wmma<Bc * Lc, Dc, Dc, 2, 4, OUT_F32>
      <<<dim3(Dc / 256, (Bc * Lc) / 32), 128, 0, stream>>>(ctxb, wob, d_out, 1.0f);
}
